// RegionalCosineSimilarityLoss_12446815224160
// MI455X (gfx1250) — compile-verified, matrix operand order checked by reading, not run on hardware
//
#include <hip/hip_runtime.h>
#include <hip/hip_bf16.h>

typedef __attribute__((ext_vector_type(16))) _Float16 v16h;
typedef __attribute__((ext_vector_type(8)))  _Float16 v8h;
typedef __attribute__((ext_vector_type(4)))  _Float16 v4h;
typedef __attribute__((ext_vector_type(8)))  float    v8f;

static constexpr int kB = 8;
static constexpr int kT = 2048;
static constexpr int kC = 512;
static constexpr int kBT = kB * kT;
static constexpr int kNeigh = 2;

// ---------------------------------------------------------------------------
// Kernel 0: zero the two accumulators (num: float, cnt: uint)
// ---------------------------------------------------------------------------
__global__ void rcs_init_kernel(float* accNum, unsigned int* accCnt) {
  *accNum = 0.0f;
  *accCnt = 0u;
}

// ---------------------------------------------------------------------------
// Kernel 1: row L2-normalize (f32 -> f16) + regions int64 -> int32
// One wave (32 lanes) per row of 512 floats; 8 waves per block.
// ---------------------------------------------------------------------------
__global__ __launch_bounds__(256)
void rcs_normalize_kernel(const float* __restrict__ x,
                          const long long* __restrict__ regions,
                          _Float16* __restrict__ xh,
                          int* __restrict__ r32) {
  const int row  = blockIdx.x * 8 + (threadIdx.x >> 5);
  const int lane = threadIdx.x & 31;
  if (row >= kBT) return;

  const float4* xr = (const float4*)(x + (size_t)row * kC);
  float4 f[4];
  float ss = 0.0f;
#pragma unroll
  for (int i = 0; i < 4; ++i) {
    f[i] = xr[i * 32 + lane];                       // coalesced 16B/lane
    ss += f[i].x * f[i].x + f[i].y * f[i].y + f[i].z * f[i].z + f[i].w * f[i].w;
  }
#pragma unroll
  for (int off = 16; off > 0; off >>= 1) ss += __shfl_xor(ss, off, 32);

  const float nrm   = sqrtf(ss);
  const float scale = 1.0f / fmaxf(nrm, 1e-8f);

  v4h* yr = (v4h*)(xh + (size_t)row * kC);
#pragma unroll
  for (int i = 0; i < 4; ++i) {
    v4h h;
    h[0] = (_Float16)(f[i].x * scale);
    h[1] = (_Float16)(f[i].y * scale);
    h[2] = (_Float16)(f[i].z * scale);
    h[3] = (_Float16)(f[i].w * scale);
    yr[i * 32 + lane] = h;
  }
  if (lane == 0) r32[row] = (int)regions[row];
}

// ---------------------------------------------------------------------------
// Kernel 2: banded Gram reduction via WMMA.
// Block = 256 threads = 8 waves arranged 4(M) x 2(N): macro tile 64x32.
// Each wave owns one 16x16 WMMA tile; K=512 swept in 16 steps of
// v_wmma_f32_16x16x32_f16. Sorted regions -> cheap band pruning.
// ---------------------------------------------------------------------------
__global__ __launch_bounds__(256)
void rcs_sim_band_kernel(const _Float16* __restrict__ xh,
                         const int* __restrict__ r32,
                         float* __restrict__ accNum,
                         unsigned int* __restrict__ accCnt) {
  const int b  = blockIdx.z;
  const int ri = blockIdx.y * 64;   // row (t) macro base
  const int cj = blockIdx.x * 32;   // col (s) macro base
  const int* rb = r32 + b * kT;

  // --- macro-tile early exits (regions sorted ascending along T) ---
  if (cj + 31 <= ri) return;                       // no s > t in tile
  {
    const int rminC = rb[cj],  rmaxC = rb[cj + 31];
    const int rminR = rb[ri],  rmaxR = rb[ri + 63];
    if (rminC - rmaxR > kNeigh) return;            // columns past band
    if (rminR - rmaxC > kNeigh) return;            // rows past band
    if (rmaxR == 0 || rmaxC == 0) return;          // fully padding
  }

  const int wave = threadIdx.x >> 5;
  const int wi   = wave & 3;                       // M sub-tile 0..3
  const int wj   = wave >> 2;                      // N sub-tile 0..1
  const int ti   = ri + wi * 16;
  const int tj   = cj + wj * 16;
  const int lane = threadIdx.x & 31;
  const int half = lane >> 4;
  const int m    = lane & 15;

  float    num = 0.0f;
  int      cnt = 0;

  bool active = (tj + 15 > ti);                    // some s > t in 16x16 tile
  if (active) {
    const int rminC = rb[tj], rmaxC = rb[tj + 15];
    const int rminR = rb[ti], rmaxR = rb[ti + 15];
    active = (rminC - rmaxR <= kNeigh) && (rminR - rmaxC <= kNeigh) &&
             (rmaxR != 0) && (rmaxC != 0);
  }

  if (active) {
    // A fragment: row M = m of tile ti; lane holds K chunks [half*8, +8) and
    // [16+half*8, +8) per ISA 16-bit A layout. B fragment mirrors it with
    // column N = m of tile tj (B column n == row tj+n of Xn).
    const _Float16* Arow = xh + ((size_t)(b * kT + ti + m)) * kC;
    const _Float16* Brow = xh + ((size_t)(b * kT + tj + m)) * kC;

    v8f c = {};
#pragma unroll
    for (int k0 = 0; k0 < kC; k0 += 32) {
      union { v16h v; v8h h[2]; } a, bt;
      a.h[0]  = *(const v8h*)(Arow + k0 + half * 8);
      a.h[1]  = *(const v8h*)(Arow + k0 + 16 + half * 8);
      bt.h[0] = *(const v8h*)(Brow + k0 + half * 8);
      bt.h[1] = *(const v8h*)(Brow + k0 + 16 + half * 8);
      c = __builtin_amdgcn_wmma_f32_16x16x32_f16(
              /*neg_a=*/false, a.v, /*neg_b=*/false, bt.v,
              /*c_mod=*/(short)0, c, /*reuse_a=*/false, /*reuse_b=*/false);
    }

    // C/D layout: lane -> N = lane%16; VGPR v -> M = half*8 + v.
    const int s  = tj + m;
    const int rs = rb[s];
#pragma unroll
    for (int v = 0; v < 8; ++v) {
      const int t  = ti + half * 8 + v;
      const int rt = rb[t];
      int d = rs - rt; d = (d < 0) ? -d : d;
      if ((s > t) && (d <= kNeigh) && (rt != 0) && (rs != 0)) {
        num += (d == 0) ? c[v] : -c[v];
        ++cnt;
      }
    }
  }

  // wave-level reduction, then one atomic per wave
#pragma unroll
  for (int off = 16; off > 0; off >>= 1) {
    num += __shfl_xor(num, off, 32);
    cnt += __shfl_xor(cnt, off, 32);
  }
  if (lane == 0) {
    if (cnt)        atomicAdd(accCnt, (unsigned int)cnt);
    if (num != 0.f) atomicAdd(accNum, num);
  }
}

// ---------------------------------------------------------------------------
// Kernel 3: finalize scalar loss
// ---------------------------------------------------------------------------
__global__ void rcs_finalize_kernel(const float* accNum,
                                    const unsigned int* accCnt,
                                    float* out) {
  out[0] = 1.0f - (*accNum) / ((float)(*accCnt) + 1e-6f);
}

// ---------------------------------------------------------------------------
// Host launcher
// ---------------------------------------------------------------------------
extern "C" void kernel_launch(void* const* d_in, const int* in_sizes, int n_in,
                              void* d_out, int out_size, void* d_ws, size_t ws_size,
                              hipStream_t stream) {
  const float*     x       = (const float*)d_in[0];
  const long long* regions = (const long long*)d_in[1];
  float*           out     = (float*)d_out;

  // workspace layout: [ xh: BT*C f16 | r32: BT int | accNum f32 | accCnt u32 ]
  const size_t xh_bytes = (size_t)kBT * kC * sizeof(_Float16);   // 16 MB
  _Float16*     xh     = (_Float16*)d_ws;
  int*          r32    = (int*)((char*)d_ws + xh_bytes);
  float*        accNum = (float*)((char*)d_ws + xh_bytes + (size_t)kBT * sizeof(int));
  unsigned int* accCnt = (unsigned int*)(accNum + 1);

  rcs_init_kernel<<<1, 1, 0, stream>>>(accNum, accCnt);

  rcs_normalize_kernel<<<kBT / 8, 256, 0, stream>>>(x, regions, xh, r32);

  dim3 grid(kT / 32, kT / 64, kB);   // (64, 32, 8)
  rcs_sim_band_kernel<<<grid, 256, 0, stream>>>(xh, r32, accNum, accCnt);

  rcs_finalize_kernel<<<1, 1, 0, stream>>>(accNum, accCnt, out);
}
